// Attention_76665166233961
// MI455X (gfx1250) — compile-verified, matrix operand order checked by reading, not run on hardware
//
#include <hip/hip_runtime.h>
#include <hip/hip_bf16.h>
#include <math.h>

// ---------------------------------------------------------------------------
// Problem constants
// ---------------------------------------------------------------------------
#define BATCH    32
#define SEQ      2048
#define ENC2     600          // ENC_HID * N_DIR
#define DECH     900          // DEC_HID
#define INDIM    1500         // DECH + ENC2
#define NPAD     928          // 900 padded to multiple of 32 (58 chunks, 29 pairs)
#define KPAD     608          // 600 padded to multiple of 32 (19 k-steps)
#define NCHUNK   58           // NPAD / 16
#define NPAIR    29           // NCHUNK / 2
#define KT       19           // KPAD / 32
#define TILE_ELEMS 512        // 32 lanes * 16 bf16 per (chunk,t) B tile

typedef __bf16 bf16_t;
typedef bf16_t v8bf  __attribute__((ext_vector_type(8)));
typedef bf16_t v16bf __attribute__((ext_vector_type(16)));
typedef float  v8f   __attribute__((ext_vector_type(8)));

__device__ __forceinline__ unsigned short f32_to_bf16_rne(float f) {
    union { float f; unsigned u; } x; x.f = f;
    unsigned u = x.u;
    u += 0x7FFFu + ((u >> 16) & 1u);      // round to nearest even
    return (unsigned short)(u >> 16);
}

// ---------------------------------------------------------------------------
// Prep 1: hb[b][o] = attn_b[o] + sum_d hidden[b,d] * attn_W[o,d]   (o < 900)
//         zero in the padded region o in [900,928)
// ---------------------------------------------------------------------------
__global__ void prep_hb_kernel(const float* __restrict__ hidden,
                               const float* __restrict__ attn_W,
                               const float* __restrict__ attn_b,
                               float* __restrict__ hb) {
    int idx = blockIdx.x * blockDim.x + threadIdx.x;
    if (idx >= BATCH * NPAD) return;
    int o = idx % NPAD;
    int b = idx / NPAD;
    float acc = 0.0f;
    if (o < DECH) {
        acc = attn_b[o];
        const float* hrow = hidden + b * DECH;
        const float* wrow = attn_W + (size_t)o * INDIM;   // Wh part: cols [0,900)
        for (int d = 0; d < DECH; ++d)
            acc = fmaf(hrow[d], wrow[d], acc);
    }
    hb[idx] = acc;
}

// ---------------------------------------------------------------------------
// Prep 2: pack We^T into WMMA B-fragment order, bf16, zero padded.
//   bpack[chunk][t][lane][j]  (chunk<58, t<19, lane<32, j<16)
//   col = lane&15, kh = (lane>>4)*8
//   j<8 : k = t*32 + kh + j        j>=8 : k = t*32 + kh + 8 + j
//   o   = chunk*16 + col
// Also writes vpad (zero padded to 928).
// ---------------------------------------------------------------------------
__global__ void prep_bpack_kernel(const float* __restrict__ attn_W,
                                  const float* __restrict__ v,
                                  unsigned short* __restrict__ bpack,
                                  float* __restrict__ vpad) {
    int idx = blockIdx.x * blockDim.x + threadIdx.x;
    if (idx < NPAD)
        vpad[idx] = (idx < DECH) ? v[idx] : 0.0f;
    if (idx >= NCHUNK * KT * TILE_ELEMS) return;
    int j     = idx & 15;
    int lane  = (idx >> 4) & 31;
    int t     = (idx >> 9) % KT;
    int chunk = idx / (KT * TILE_ELEMS);
    int col   = lane & 15;
    int kh    = (lane >> 4) << 3;
    int k     = t * 32 + kh + ((j < 8) ? j : (8 + j));
    int o     = chunk * 16 + col;
    float val = (o < DECH && k < ENC2) ? attn_W[(size_t)o * INDIM + DECH + k] : 0.0f;
    bpack[idx] = f32_to_bf16_rne(val);
}

// ---------------------------------------------------------------------------
// Main fused kernel: scores[b,s] = sum_o v[o] * tanh(eproj[b,s,o] + hb[b,o])
// Block = 64 threads (2 waves), 16 rows of flattened (b*S+s) per wave.
// A tile: f32 -> bf16 once into LDS, then 19 v16bf fragments in registers.
// N loop: 29 pairs of chunks, two independent WMMA accumulator chains.
// ---------------------------------------------------------------------------
__global__ void __launch_bounds__(64)
attn_scores_kernel(const float* __restrict__ enc,            // [B*S, 600]
                   const unsigned short* __restrict__ bpack, // packed B frags
                   const float* __restrict__ hb,             // [B, 928]
                   const float* __restrict__ vpad,           // [928]
                   float* __restrict__ scores) {             // [B*S]
    __shared__ unsigned short Atile[2][16 * KPAD];           // 38,912 bytes

    const int wid  = threadIdx.x >> 5;
    const int lane = threadIdx.x & 31;
    const int col  = lane & 15;                 // A row / C column
    const int kh   = (lane >> 4) << 3;          // K sub-offset per lane half
    const int mh   = (lane >> 4) << 3;          // C row group per lane half

    const int gRowBase = blockIdx.x * 32 + wid * 16;        // 16 rows per wave
    const int b = gRowBase >> 11;                            // /2048, uniform

    // ---- stage 16x608 bf16 A tile into LDS (convert once) ----
    unsigned short* myA = &Atile[wid][0];
    const float* encBase = enc + (size_t)gRowBase * ENC2;
    for (int p = lane; p < 16 * (KPAD / 2); p += 32) {
        int row = p / (KPAD / 2);
        int kp  = (p % (KPAD / 2)) * 2;
        float f0 = (kp     < ENC2) ? encBase[(size_t)row * ENC2 + kp]     : 0.0f;
        float f1 = (kp + 1 < ENC2) ? encBase[(size_t)row * ENC2 + kp + 1] : 0.0f;
        ((unsigned int*)myA)[p] =
            (unsigned)f32_to_bf16_rne(f0) | ((unsigned)f32_to_bf16_rne(f1) << 16);
    }
    __syncthreads();

    // ---- load the 19 A fragments into registers (documented 16-bit layout) ----
    const unsigned short* Arow = myA + col * KPAD;
    v16bf afrag[KT];
#pragma unroll
    for (int t = 0; t < KT; ++t) {
        v8bf a_lo = *(const v8bf*)(Arow + t * 32 + kh);
        v8bf a_hi = *(const v8bf*)(Arow + t * 32 + kh + 16);
        afrag[t] = __builtin_shufflevector(a_lo, a_hi,
                     0,1,2,3,4,5,6,7,8,9,10,11,12,13,14,15);
    }

    float sAcc[8] = {0.f, 0.f, 0.f, 0.f, 0.f, 0.f, 0.f, 0.f};

    for (int p = 0; p < NPAIR; ++p) {
        const int chunk0 = 2 * p;
        v8f c0 = {0.f, 0.f, 0.f, 0.f, 0.f, 0.f, 0.f, 0.f};
        v8f c1 = {0.f, 0.f, 0.f, 0.f, 0.f, 0.f, 0.f, 0.f};
        // coalesced B fragments: lane's 16 bf16 are 32 contiguous bytes
        const unsigned short* b0p = bpack + (size_t)(chunk0    ) * KT * TILE_ELEMS + lane * 16;
        const unsigned short* b1p = bpack + (size_t)(chunk0 + 1) * KT * TILE_ELEMS + lane * 16;

#pragma unroll
        for (int t = 0; t < KT; ++t) {
            v16bf bm0 = *(const v16bf*)(b0p + t * TILE_ELEMS);
            v16bf bm1 = *(const v16bf*)(b1p + t * TILE_ELEMS);
            c0 = __builtin_amdgcn_wmma_f32_16x16x32_bf16(
                     false, afrag[t], false, bm0, (short)0, c0, false, false);
            c1 = __builtin_amdgcn_wmma_f32_16x16x32_bf16(
                     false, afrag[t], false, bm1, (short)0, c1, false, false);
        }

        // epilogue: C layout — c[r] is (m = mh+r, n = col) of each N chunk
        const int o0 = chunk0 * 16 + col;
        const float hb0 = hb[b * NPAD + o0];
        const float hb1 = hb[b * NPAD + o0 + 16];
        const float vv0 = vpad[o0];
        const float vv1 = vpad[o0 + 16];
#pragma unroll
        for (int r = 0; r < 8; ++r) {
            sAcc[r] += vv0 * tanhf(c0[r] + hb0);
            sAcc[r] += vv1 * tanhf(c1[r] + hb1);
        }
    }

    // reduce over N (the 16 lanes of each half hold the 16 columns)
#pragma unroll
    for (int r = 0; r < 8; ++r) {
        float x = sAcc[r];
        x += __shfl_xor(x, 1);
        x += __shfl_xor(x, 2);
        x += __shfl_xor(x, 4);
        x += __shfl_xor(x, 8);
        sAcc[r] = x;
    }

    if (col == 0) {
#pragma unroll
        for (int r = 0; r < 8; ++r)
            scores[gRowBase + mh + r] = sAcc[r];
    }
}

// ---------------------------------------------------------------------------
// Softmax over S=2048, in place on d_out. One block per batch row.
// ---------------------------------------------------------------------------
__global__ void __launch_bounds__(256)
softmax_kernel(float* __restrict__ out) {
    __shared__ float red[256];
    float* row = out + (size_t)blockIdx.x * SEQ;
    const int t = threadIdx.x;

    float mx = -INFINITY;
    float vals[8];
#pragma unroll
    for (int j = 0; j < 8; ++j) {
        vals[j] = row[t + j * 256];
        mx = fmaxf(mx, vals[j]);
    }
    red[t] = mx;
    __syncthreads();
    for (int s = 128; s > 0; s >>= 1) {
        if (t < s) red[t] = fmaxf(red[t], red[t + s]);
        __syncthreads();
    }
    mx = red[0];
    __syncthreads();

    float sum = 0.f;
#pragma unroll
    for (int j = 0; j < 8; ++j) {
        vals[j] = __expf(vals[j] - mx);
        sum += vals[j];
    }
    red[t] = sum;
    __syncthreads();
    for (int s = 128; s > 0; s >>= 1) {
        if (t < s) red[t] += red[t + s];
        __syncthreads();
    }
    const float inv = 1.0f / red[0];
#pragma unroll
    for (int j = 0; j < 8; ++j)
        row[t + j * 256] = vals[j] * inv;
}

// ---------------------------------------------------------------------------
// Launch
// ---------------------------------------------------------------------------
extern "C" void kernel_launch(void* const* d_in, const int* in_sizes, int n_in,
                              void* d_out, int out_size, void* d_ws, size_t ws_size,
                              hipStream_t stream) {
    const float* hidden = (const float*)d_in[0];   // [32, 900]
    const float* enc    = (const float*)d_in[1];   // [32, 2048, 600]
    const float* attn_W = (const float*)d_in[2];   // [900, 1500]
    const float* attn_b = (const float*)d_in[3];   // [900]
    const float* v      = (const float*)d_in[4];   // [900]
    float* out = (float*)d_out;                    // [32, 2048]

    char* ws = (char*)d_ws;
    // hb: 32*928 f32 = 118,784 B (256-aligned size)
    float* hb = (float*)ws;
    // vpad: 928 f32 = 3,712 B at 118784; next region rounded to 122624
    float* vpad = (float*)(ws + 118784);
    // bpack: 58*19*512 bf16 = 1,128,448 B
    unsigned short* bpack = (unsigned short*)(ws + 122624);

    {
        int n = BATCH * NPAD;
        prep_hb_kernel<<<(n + 255) / 256, 256, 0, stream>>>(hidden, attn_W, attn_b, hb);
    }
    {
        int n = NCHUNK * KT * TILE_ELEMS;
        prep_bpack_kernel<<<(n + 255) / 256, 256, 0, stream>>>(attn_W, v, bpack, vpad);
    }
    {
        int nblocks = (BATCH * SEQ) / 32;          // 2048 blocks, 2 waves each
        attn_scores_kernel<<<nblocks, 64, 0, stream>>>(enc, bpack, hb, vpad, out);
    }
    softmax_kernel<<<BATCH, 256, 0, stream>>>(out);
}
